// MultiHeadAttention_35201551958758
// MI455X (gfx1250) — matrix operand order repaired and checked by reading, not compiled, not run on hardware
//
#include <hip/hip_runtime.h>
#include <hip/hip_bf16.h>

typedef __attribute__((ext_vector_type(16))) _Float16 v16h;
typedef __attribute__((ext_vector_type(8)))  float    v8f;

#define DEV static __device__ __forceinline__

// ---------------- constants ----------------
#define BATCH   4
#define SEQ     2048
#define DMODEL  512
#define NHEADS  8
#define DEPTH   64
#define NEG_INF -1000000000.0f

DEV v8f wmma_f16(v16h a, v16h b, v8f c) {
    return __builtin_amdgcn_wmma_f32_16x16x32_f16(false, a, false, b, (short)0, c, false, false);
}

// A-fragment (16x32 f16): lane holds row (lane&15); half i -> K = hi*16 + i
DEV v16h load_a_f32row(const float* row, int kc, int hi) {
    v16h a;
    const float* p0 = row + kc + hi * 16;      // K = kc+hi*16 .. +7
    const float* p1 = p0 + 8;                  // K = kc+hi*16+8 .. +15
#pragma unroll
    for (int i = 0; i < 8; ++i) { a[i] = (_Float16)p0[i]; a[8 + i] = (_Float16)p1[i]; }
    return a;
}
DEV v16h load_a_f16row(const _Float16* row, int kc, int hi) {
    v16h a;
    const _Float16* p0 = row + kc + hi * 16;
    const _Float16* p1 = p0 + 8;
#pragma unroll
    for (int i = 0; i < 8; ++i) { a[i] = p0[i]; a[8 + i] = p1[i]; }
    return a;
}

DEV void lds_fence() {
    __builtin_amdgcn_wave_barrier();
    asm volatile("s_wait_dscnt 0" ::: "memory");
}

DEV void rowreduce_max8(float* v) {
#pragma unroll
    for (int d = 1; d < 16; d <<= 1) {
#pragma unroll
        for (int r = 0; r < 8; ++r) v[r] = fmaxf(v[r], __shfl_xor(v[r], d, 32));
    }
}
DEV void rowreduce_sum8(float* v) {
#pragma unroll
    for (int d = 1; d < 16; d <<= 1) {
#pragma unroll
        for (int r = 0; r < 8; ++r) v[r] += __shfl_xor(v[r], d, 32);
    }
}

// ---------------------------------------------------------------------------
// Weight transpose + f16 convert: Wt[col][k] = (f16) W[k][col]  (512x512)
// ---------------------------------------------------------------------------
__global__ __launch_bounds__(256) void wt_kernel(const float* __restrict__ W,
                                                 _Float16* __restrict__ Wt) {
    const int idx = blockIdx.x * 256 + threadIdx.x;   // 0..262143
    const int col = idx >> 9;
    const int k   = idx & 511;
    Wt[(size_t)col * DMODEL + k] = (_Float16)W[(size_t)k * DMODEL + col];
}

// ---------------------------------------------------------------------------
// Mask bit-pack: mbits[b][s][w] bit j = (mask[b][s][w*32+j] != 0)
// One wave packs 32 ints -> 1 word via shuffle-OR reduce. 32 words per wave.
// ---------------------------------------------------------------------------
__global__ __launch_bounds__(128) void mask_pack_kernel(const int* __restrict__ mask,
                                                        unsigned* __restrict__ mbits) {
    const int wid  = threadIdx.x >> 5;
    const int lane = threadIdx.x & 31;
    size_t w0 = ((size_t)blockIdx.x * 4 + wid) * 32;
    for (int j = 0; j < 32; ++j) {
        const size_t w = w0 + j;
        unsigned bit = (mask[w * 32 + lane] != 0) ? (1u << lane) : 0u;
#pragma unroll
        for (int d = 1; d < 32; d <<= 1) bit |= __shfl_xor(bit, d, 32);
        if (lane == 0) mbits[w] = bit;
    }
}

// ---------------------------------------------------------------------------
// QKV projection: Y = X(f32)@W + b, f16 out.  Wt is f16 column-major.
// One wave computes a 64x16 output strip (4 M-tiles share each B-fragment).
// VT=false: head-split [B,H,S,64].  VT=true: transposed per head [B,H,64,S].
// ---------------------------------------------------------------------------
template <bool VT>
__global__ __launch_bounds__(128) void proj_qkv_kernel(const float* __restrict__ X,
                                                       const _Float16* __restrict__ Wt,
                                                       const float* __restrict__ bias,
                                                       _Float16* __restrict__ Y) {
    const int wave = blockIdx.x * 4 + (threadIdx.x >> 5);
    const int mt = wave >> 5;            // 0..127 (64-row supertile)
    const int nt = wave & 31;            // 0..31
    const int lane = threadIdx.x & 31;
    const int hi = lane >> 4, ln = lane & 15;
    const int m0 = mt * 64;
    const int col = nt * 16 + ln;

    const float bv = bias[col];
    v8f acc[4];
#pragma unroll
    for (int t = 0; t < 4; ++t)
#pragma unroll
        for (int r = 0; r < 8; ++r) acc[t][r] = bv;

    const _Float16* wcol = Wt + (size_t)col * DMODEL;
    const float* arow = X + (size_t)(m0 + ln) * DMODEL;

    for (int kc = 0; kc < DMODEL; kc += 32) {
        const v16h bf = *(const v16h*)(wcol + kc + hi * 16);
#pragma unroll
        for (int t = 0; t < 4; ++t) {
            v16h a = load_a_f32row(arow + (size_t)t * 16 * DMODEL, kc, hi);
            acc[t] = wmma_f16(a, bf, acc[t]);
        }
    }

    const int h = col >> 6, d = col & 63;
#pragma unroll
    for (int t = 0; t < 4; ++t) {
#pragma unroll
        for (int r = 0; r < 8; ++r) {
            const int row = m0 + t * 16 + r + hi * 8;     // global token index
            const int bb = row >> 11, s = row & 2047;
            if (VT)
                Y[(((size_t)bb * NHEADS + h) * DEPTH + d) * SEQ + s] = (_Float16)acc[t][r];
            else
                Y[(((size_t)bb * NHEADS + h) * SEQ + s) * DEPTH + d] = (_Float16)acc[t][r];
        }
    }
}

// ---------------------------------------------------------------------------
// Output projection: out(f32) = ctx(f16)@Wo + bo.  Wt f16 column-major.
// ---------------------------------------------------------------------------
__global__ __launch_bounds__(128) void proj_out_kernel(const _Float16* __restrict__ Xh,
                                                       const _Float16* __restrict__ Wt,
                                                       const float* __restrict__ bias,
                                                       float* __restrict__ out) {
    const int wave = blockIdx.x * 4 + (threadIdx.x >> 5);
    const int mt = wave >> 5;
    const int nt = wave & 31;
    const int lane = threadIdx.x & 31;
    const int hi = lane >> 4, ln = lane & 15;
    const int m0 = mt * 64;
    const int col = nt * 16 + ln;

    const float bv = bias[col];
    v8f acc[4];
#pragma unroll
    for (int t = 0; t < 4; ++t)
#pragma unroll
        for (int r = 0; r < 8; ++r) acc[t][r] = bv;

    const _Float16* wcol = Wt + (size_t)col * DMODEL;
    const _Float16* arow = Xh + (size_t)(m0 + ln) * DMODEL;

    for (int kc = 0; kc < DMODEL; kc += 32) {
        const v16h bf = *(const v16h*)(wcol + kc + hi * 16);
#pragma unroll
        for (int t = 0; t < 4; ++t) {
            v16h a = load_a_f16row(arow + (size_t)t * 16 * DMODEL, kc, hi);
            acc[t] = wmma_f16(a, bf, acc[t]);
        }
    }

#pragma unroll
    for (int t = 0; t < 4; ++t) {
#pragma unroll
        for (int r = 0; r < 8; ++r) {
            const int row = m0 + t * 16 + r + hi * 8;
            out[(size_t)row * DMODEL + col] = acc[t][r];
        }
    }
}

// ---------------------------------------------------------------------------
// Fused flash attention. One wave: 16 query rows of one (b,h).
// Per 32-key chunk: 4 WMMAs QK^T (contiguous K loads), bitmask (1 dword load),
// online softmax, P via LDS transpose, 4 WMMAs P*V (contiguous V^T loads).
// ---------------------------------------------------------------------------
__global__ __launch_bounds__(128) void attn_kernel(const _Float16* __restrict__ qh,
                                                   const _Float16* __restrict__ kh,
                                                   const _Float16* __restrict__ vt,
                                                   const unsigned* __restrict__ mbits,
                                                   _Float16* __restrict__ ctx) {
    __shared__ _Float16 Pbuf[4][16 * 32];   // per-wave 16x32 P tile (f16)

    const int wid = threadIdx.x >> 5;
    const int wave = blockIdx.x * 4 + wid;
    const int qt = wave & 127;              // query tile (S/16 = 128)
    const int h  = (wave >> 7) & 7;
    const int b  = wave >> 10;
    const int lane = threadIdx.x & 31;
    const int hi = lane >> 4, ln = lane & 15;
    const int q0 = qt * 16;

    const size_t headOff = ((size_t)b * NHEADS + h) * SEQ * DEPTH;
    const _Float16* qbase  = qh + headOff;
    const _Float16* kbase  = kh + headOff;
    const _Float16* vtbase = vt + headOff;                     // [64][SEQ]
    const unsigned* mwrow  = mbits + ((size_t)b * SEQ + q0 + ln) * (SEQ / 32);
    _Float16* P = Pbuf[wid];

    // Q A-fragments (depth 0-31 and 32-63), reused for all key chunks
    const _Float16* qrow = qbase + (size_t)(q0 + ln) * DEPTH;
    const v16h qa0 = load_a_f16row(qrow, 0, hi);
    const v16h qa1 = load_a_f16row(qrow, 32, hi);

    float mstat[8], lstat[8];
#pragma unroll
    for (int r = 0; r < 8; ++r) { mstat[r] = -1e30f; lstat[r] = 0.0f; }
    v8f o0 = {}, o1 = {}, o2 = {}, o3 = {};

    for (int kc = 0; kc < SEQ; kc += 32) {
        // ---- scores: two 16x16 tiles (keys kc..+15 and kc+16..+31) ----
        v8f s0 = {}, s1 = {};
        {
            const _Float16* krow0 = kbase + (size_t)(kc + ln) * DEPTH;
            const _Float16* krow1 = krow0 + 16 * DEPTH;
            v16h kb = *(const v16h*)(krow0 + hi * 16);
            s0 = wmma_f16(qa0, kb, s0);
            kb = *(const v16h*)(krow0 + 32 + hi * 16);
            s0 = wmma_f16(qa1, kb, s0);
            kb = *(const v16h*)(krow1 + hi * 16);
            s1 = wmma_f16(qa0, kb, s1);
            kb = *(const v16h*)(krow1 + 32 + hi * 16);
            s1 = wmma_f16(qa1, kb, s1);
        }

        // ---- scale + mask (lane ln loaded the word for q-row ln) ----
        const unsigned wv = mwrow[kc >> 5];
#pragma unroll
        for (int r = 0; r < 8; ++r) {
            const unsigned wr = __shfl(wv, r + hi * 8, 32);  // word of row r+hi*8
            const float v0 = s0[r] * 0.125f;                 // 1/sqrt(64)
            const float v1 = s1[r] * 0.125f;
            s0[r] = ((wr >> ln) & 1u)        ? v0 : NEG_INF;
            s1[r] = ((wr >> (16 + ln)) & 1u) ? v1 : NEG_INF;
        }

        // ---- online softmax over the 32-key chunk ----
        float rmax[8];
#pragma unroll
        for (int r = 0; r < 8; ++r) rmax[r] = fmaxf(s0[r], s1[r]);
        rowreduce_max8(rmax);

        float scale[8], rsum[8], p0[8], p1[8];
#pragma unroll
        for (int r = 0; r < 8; ++r) {
            const float mo = mstat[r];
            const float mn = fmaxf(mo, rmax[r]);
            mstat[r] = mn;
            scale[r] = __expf(mo - mn);
            p0[r] = __expf(s0[r] - mn);
            p1[r] = __expf(s1[r] - mn);
            rsum[r] = p0[r] + p1[r];
        }
        rowreduce_sum8(rsum);
#pragma unroll
        for (int r = 0; r < 8; ++r) {
            lstat[r] = lstat[r] * scale[r] + rsum[r];
            o0[r] *= scale[r]; o1[r] *= scale[r];
            o2[r] *= scale[r]; o3[r] *= scale[r];
        }

        // ---- stash P (C layout) into LDS, reload as A-fragment ----
#pragma unroll
        for (int r = 0; r < 8; ++r) {
            const int prow = r + hi * 8;
            P[prow * 32 + ln]      = (_Float16)p0[r];
            P[prow * 32 + 16 + ln] = (_Float16)p1[r];
        }
        lds_fence();
        v16h pa;
        {
            const _Float16* pr0 = P + ln * 32 + hi * 16;      // klocal hi*16 .. +7
            const _Float16* pr1 = pr0 + 8;
#pragma unroll
            for (int i = 0; i < 8; ++i) { pa[i] = pr0[i]; pa[8 + i] = pr1[i]; }
        }
        lds_fence();   // keep next-iteration LDS writes ordered after reads

        // ---- O += P @ V   (V^T rows contiguous: one v16h per fragment) ----
#pragma unroll
        for (int dt = 0; dt < 4; ++dt) {
            const v16h vb = *(const v16h*)(vtbase + (size_t)(dt * 16 + ln) * SEQ + kc + hi * 16);
            v8f* op = (dt == 0) ? &o0 : (dt == 1) ? &o1 : (dt == 2) ? &o2 : &o3;
            *op = wmma_f16(pa, vb, *op);
        }
    }

    // ---- normalize and write ctx [B,S,512] (f16) ----
    float inv[8];
#pragma unroll
    for (int r = 0; r < 8; ++r) inv[r] = 1.0f / lstat[r];
#pragma unroll
    for (int dt = 0; dt < 4; ++dt) {
        const v8f& o = (dt == 0) ? o0 : (dt == 1) ? o1 : (dt == 2) ? o2 : o3;
        const int col = h * DEPTH + dt * 16 + ln;
#pragma unroll
        for (int r = 0; r < 8; ++r) {
            const int row = q0 + r + hi * 8;
            ctx[((size_t)b * SEQ + row) * DMODEL + col] = (_Float16)(o[r] * inv[r]);
        }
    }
}

// ---------------------------------------------------------------------------
extern "C" void kernel_launch(void* const* d_in, const int* in_sizes, int n_in,
                              void* d_out, int out_size, void* d_ws, size_t ws_size,
                              hipStream_t stream) {
    const float* Q  = (const float*)d_in[0];
    const float* K  = (const float*)d_in[1];
    const float* V  = (const float*)d_in[2];
    const int* mask = (const int*)d_in[3];
    const float* Wq = (const float*)d_in[4];
    const float* bq = (const float*)d_in[5];
    const float* Wk = (const float*)d_in[6];
    const float* bk = (const float*)d_in[7];
    const float* Wv = (const float*)d_in[8];
    const float* bv = (const float*)d_in[9];
    const float* Wo = (const float*)d_in[10];
    const float* bo = (const float*)d_in[11];

    const size_t HEAD_BYTES = (size_t)BATCH * NHEADS * SEQ * DEPTH * sizeof(_Float16); // 8 MB
    const size_t WT_BYTES   = (size_t)DMODEL * DMODEL * sizeof(_Float16);              // 0.5 MB
    char* ws = (char*)d_ws;
    _Float16* qh   = (_Float16*)(ws + 0 * HEAD_BYTES);
    _Float16* kh   = (_Float16*)(ws + 1 * HEAD_BYTES);
    _Float16* vt   = (_Float16*)(ws + 2 * HEAD_BYTES);
    _Float16* ctx  = (_Float16*)(ws + 3 * HEAD_BYTES);
    char* wbase    = ws + 4 * HEAD_BYTES;
    _Float16* wtq  = (_Float16*)(wbase + 0 * WT_BYTES);
    _Float16* wtk  = (_Float16*)(wbase + 1 * WT_BYTES);
    _Float16* wtv  = (_Float16*)(wbase + 2 * WT_BYTES);
    _Float16* wto  = (_Float16*)(wbase + 3 * WT_BYTES);
    unsigned* mbits = (unsigned*)(wbase + 4 * WT_BYTES);   // 2 MB

    // 1) weight transposes (f32 -> f16 column-major)
    wt_kernel<<<1024, 256, 0, stream>>>(Wq, wtq);
    wt_kernel<<<1024, 256, 0, stream>>>(Wk, wtk);
    wt_kernel<<<1024, 256, 0, stream>>>(Wv, wtv);
    wt_kernel<<<1024, 256, 0, stream>>>(Wo, wto);

    // 2) mask bit-pack: 4*2048*64 = 524288 words, 128 words per block
    mask_pack_kernel<<<4096, 128, 0, stream>>>(mask, mbits);

    // 3) projections: (8192/64)*(512/16) = 4096 waves, 4 per block
    const int projBlocks = 1024;
    proj_qkv_kernel<false><<<projBlocks, 128, 0, stream>>>(Q, wtq, bq, qh);
    proj_qkv_kernel<false><<<projBlocks, 128, 0, stream>>>(K, wtk, bk, kh);
    proj_qkv_kernel<true ><<<projBlocks, 128, 0, stream>>>(V, wtv, bv, vt);

    // 4) attention: B*H*(S/16) = 4096 waves, 4 per block
    attn_kernel<<<1024, 128, 0, stream>>>(qh, kh, vt, mbits, ctx);

    // 5) output projection
    proj_out_kernel<<<projBlocks, 128, 0, stream>>>(ctx, wto, bo, (float*)d_out);
}
